// NTXentLoss_8169027797001
// MI455X (gfx1250) — matrix operand order repaired and checked by reading, not compiled, not run on hardware
//
#include <hip/hip_runtime.h>
#include <hip/hip_bf16.h>
#include <cmath>

typedef __attribute__((ext_vector_type(16))) _Float16 v16h;
typedef __attribute__((ext_vector_type(8)))  _Float16 v8h;
typedef __attribute__((ext_vector_type(8)))  float    v8f;

#define NROWS 8192
#define DDIM  1024
#define HALF_N 4096
#define INV_T 2.0f          // 1 / TEMPERATURE (0.5)
#define MAXLOGIT 2.0f       // |cos sim| <= 1 -> sim/T in [-2,2]; fixed softmax max
#define MTILES 4            // 4 x 16 = 64-row panel per block (B reuse x4 over M)
#define MROWS (MTILES * 16)
#define NTILES 2            // j-tiles per wave iteration (A-fragment reuse x2)
// Interleaved LDS A layout: group g = cl (0..15) holds the 4 sub-panel rows
// with that intra-tile row index, at p*1032 halves within the group.
// Group stride 4232 halves = 2116 dwords == 4 (mod 64) -> cl lanes hit
// distinct bank groups; fragment offsets (p*2064 + k*2 + {0,32} B) all fit
// the 16-bit DS immediate-offset field.
#define LDSG 4232

// ---------------------------------------------------------------------------
// Kernel 1: row L2-normalize  x[row]/max(||x||,eps)  ->  f16 matrix Xh
// ---------------------------------------------------------------------------
__global__ void ntx_normalize(const float* __restrict__ zi,
                              const float* __restrict__ zj,
                              _Float16* __restrict__ Xh) {
    const int row = blockIdx.x;
    const float* src = (row < HALF_N) ? (zi + (size_t)row * DDIM)
                                      : (zj + (size_t)(row - HALF_N) * DDIM);
    __shared__ float red[256];
    float v[4];
    float ss = 0.0f;
#pragma unroll
    for (int it = 0; it < 4; ++it) {
        v[it] = src[threadIdx.x + it * 256];
        ss += v[it] * v[it];
    }
    red[threadIdx.x] = ss;
    __syncthreads();
    for (int s = 128; s > 0; s >>= 1) {
        if (threadIdx.x < s) red[threadIdx.x] += red[threadIdx.x + s];
        __syncthreads();
    }
    const float inv = 1.0f / fmaxf(sqrtf(red[0]), 1e-8f);
    _Float16* dst = Xh + (size_t)row * DDIM;
#pragma unroll
    for (int it = 0; it < 4; ++it)
        dst[threadIdx.x + it * 256] = (_Float16)(v[it] * inv);
}

// ---------------------------------------------------------------------------
// Kernel 2: f32 target similarities t[i] = dot(xn[i], xn[i^1]).
// ---------------------------------------------------------------------------
__global__ void ntx_pair_dot(const _Float16* __restrict__ Xh,
                             float* __restrict__ tvec) {
    const int p = blockIdx.x;
    const _Float16* r0 = Xh + (size_t)(2 * p) * DDIM;
    const _Float16* r1 = r0 + DDIM;
    __shared__ float red[256];
    float acc = 0.0f;
#pragma unroll
    for (int it = 0; it < 4; ++it) {
        const int d = threadIdx.x + it * 256;
        acc += (float)r0[d] * (float)r1[d];
    }
    red[threadIdx.x] = acc;
    __syncthreads();
    for (int s = 128; s > 0; s >>= 1) {
        if (threadIdx.x < s) red[threadIdx.x] += red[threadIdx.x + s];
        __syncthreads();
    }
    if (threadIdx.x == 0) {
        tvec[2 * p]     = red[0];
        tvec[2 * p + 1] = red[0];
    }
}

// ---------------------------------------------------------------------------
// Kernel 3: fused Gram-GEMM (WMMA f16 -> f32 acc) + fixed-max sum-exp.
// One block = one 64-row panel. Each wave iterates PAIRS of j-tiles:
// per k-step: 4 global_load_b128 (2 B fragments) + 8 ds_load_b128
// (4 A fragments, immediate offsets) + 8 independent WMMAs.
// s += exp(sim/T - 2) branch-free; diagonal zeroed by select.
// ---------------------------------------------------------------------------
__global__ void __launch_bounds__(256)
ntx_gram_lse(const _Float16* __restrict__ Xh,
             const float* __restrict__ tvec,
             float* __restrict__ rowval) {
    __shared__ _Float16 ldsA[16 * LDSG];   // ~132 KB (<= 320 KB/WGP)
    __shared__ float red[8 * MROWS];       // per-wave row partial sums

    const int i0   = blockIdx.x * MROWS;
    const int lane = threadIdx.x & 31;
    const int wave = threadIdx.x >> 5;
    const int hi   = lane >> 4;    // half-wave selector
    const int cl   = lane & 15;    // row (A) / column (B) index within tile

    // Stage A panel, interleaved: global row r -> group (r&15), slot (r>>4).
    for (int c = threadIdx.x; c < MROWS * (DDIM / 8); c += 256) {
        const int r  = c >> 7;
        const int dc = (c & 127) * 8;
        *(v8h*)(ldsA + (r & 15) * LDSG + (r >> 4) * 1032 + dc) =
            *(const v8h*)(Xh + (size_t)(i0 + r) * DDIM + dc);
    }
    __syncthreads();

    float s[MTILES * 8];
#pragma unroll
    for (int q = 0; q < MTILES * 8; ++q) s[q] = 0.0f;

    const _Float16* aBase = ldsA + cl * LDSG + hi * 8;

    for (int jp = wave; jp < NROWS / (16 * NTILES); jp += 8) {
        const int j0 = jp * (16 * NTILES);
        const _Float16* gB0 = Xh + (size_t)(j0 + cl) * DDIM + hi * 8;
        const _Float16* gB1 = Xh + (size_t)(j0 + 16 + cl) * DDIM + hi * 8;

        v8f acc[NTILES][MTILES];
#pragma unroll
        for (int t = 0; t < NTILES; ++t)
#pragma unroll
            for (int p = 0; p < MTILES; ++p)
                acc[t][p] = (v8f){0.f, 0.f, 0.f, 0.f, 0.f, 0.f, 0.f, 0.f};

#pragma unroll 4
        for (int k = 0; k < DDIM; k += 32) {
            // B fragments (32x16 f16): lane n holds column n (= row j+n of X),
            // K {k..k+7, k+16..k+23} (low lanes) / +8 within each chunk (high lanes).
            v8h b0lo = *(const v8h*)(gB0 + k);
            v8h b0hi = *(const v8h*)(gB0 + k + 16);
            v8h b1lo = *(const v8h*)(gB1 + k);
            v8h b1hi = *(const v8h*)(gB1 + k + 16);
            v16h b0 = __builtin_shufflevector(b0lo, b0hi, 0,1,2,3,4,5,6,7,
                                                          8,9,10,11,12,13,14,15);
            v16h b1 = __builtin_shufflevector(b1lo, b1hi, 0,1,2,3,4,5,6,7,
                                                          8,9,10,11,12,13,14,15);
#pragma unroll
            for (int p = 0; p < MTILES; ++p) {
                // A fragment (16x32 f16): lane<16 -> K {k..k+7, k+16..k+23},
                //                         lane>=16 -> +8 within each chunk.
                const _Float16* ap = aBase + p * 1032 + k;
                v8h alo = *(const v8h*)(ap);
                v8h ahi = *(const v8h*)(ap + 16);
                v16h a = __builtin_shufflevector(alo, ahi, 0,1,2,3,4,5,6,7,
                                                           8,9,10,11,12,13,14,15);
                acc[0][p] = __builtin_amdgcn_wmma_f32_16x16x32_f16(
                                false, a, false, b0, (short)0, acc[0][p], false, false);
                acc[1][p] = __builtin_amdgcn_wmma_f32_16x16x32_f16(
                                false, a, false, b1, (short)0, acc[1][p], false, false);
            }
        }
        // Branch-free fixed-max sum-exp; zero the diagonal via select.
#pragma unroll
        for (int t = 0; t < NTILES; ++t) {
            const int gcol = j0 + t * 16 + cl;
#pragma unroll
            for (int p = 0; p < MTILES; ++p) {
#pragma unroll
                for (int r = 0; r < 8; ++r) {
                    const int grow = i0 + p * 16 + 8 * hi + r;
                    const float e = __expf(acc[t][p][r] * INV_T - MAXLOGIT);
                    s[p * 8 + r] += (grow == gcol) ? 0.0f : e;
                }
            }
        }
    }

    // Sum across the 16 lanes sharing each row (xor 1,2,4,8), then one lane
    // per half-wave publishes its 32 row partials.
#pragma unroll
    for (int q = 0; q < MTILES * 8; ++q) {
        float v = s[q];
        v += __shfl_xor(v, 1, 32);
        v += __shfl_xor(v, 2, 32);
        v += __shfl_xor(v, 4, 32);
        v += __shfl_xor(v, 8, 32);
        if (cl == 0) {
            const int row_local = (q >> 3) * 16 + 8 * hi + (q & 7);
            red[wave * MROWS + row_local] = v;
        }
    }
    __syncthreads();

    if (threadIdx.x < MROWS) {
        const int r = threadIdx.x;
        float S = 0.0f;
#pragma unroll
        for (int w = 0; w < 8; ++w) S += red[w * MROWS + r];
        const float lse = MAXLOGIT + logf(S);
        rowval[i0 + r] = lse - INV_T * tvec[i0 + r];
    }
}

// ---------------------------------------------------------------------------
// Kernel 4: loss = mean(rowval)
// ---------------------------------------------------------------------------
__global__ void ntx_final(const float* __restrict__ rowval, float* __restrict__ out) {
    __shared__ float red[256];
    float acc = 0.0f;
    for (int i = threadIdx.x; i < NROWS; i += 256) acc += rowval[i];
    red[threadIdx.x] = acc;
    __syncthreads();
    for (int s = 128; s > 0; s >>= 1) {
        if (threadIdx.x < s) red[threadIdx.x] += red[threadIdx.x + s];
        __syncthreads();
    }
    if (threadIdx.x == 0) out[0] = red[0] * (1.0f / (float)NROWS);
}

// ---------------------------------------------------------------------------
extern "C" void kernel_launch(void* const* d_in, const int* in_sizes, int n_in,
                              void* d_out, int out_size, void* d_ws, size_t ws_size,
                              hipStream_t stream) {
    (void)in_sizes; (void)n_in; (void)out_size; (void)ws_size;
    const float* zi = (const float*)d_in[0];
    const float* zj = (const float*)d_in[1];

    // Workspace layout: [ Xh f16 N*D | tvec f32 N | rowval f32 N ]  (~16.1 MB)
    _Float16* Xh   = (_Float16*)d_ws;
    float*    tvec = (float*)((char*)d_ws + (size_t)NROWS * DDIM * sizeof(_Float16));
    float*    rowv = tvec + NROWS;
    float*    out  = (float*)d_out;

    ntx_normalize<<<NROWS, 256, 0, stream>>>(zi, zj, Xh);
    ntx_pair_dot <<<NROWS / 2, 256, 0, stream>>>(Xh, tvec);
    ntx_gram_lse <<<NROWS / MROWS, 256, 0, stream>>>(Xh, tvec, rowv);
    ntx_final    <<<1, 256, 0, stream>>>(rowv, out);
}
